// CausalSelfAttention_48000554500198
// MI455X (gfx1250) — matrix operand order repaired and checked by reading, not compiled, not run on hardware
//
#include <hip/hip_runtime.h>
#include <stdint.h>

// ---------------------------------------------------------------------------
// CDNA5 (gfx1250) causal self-attention: bf16 WMMA GEMMs + flash attention,
// with Tensor Data Mover (TDM) tile DMA into LDS and double buffering.
// B=2, T=4096, C=768, H=12, D=64 (hardcoded; shapes fixed by reference).
// ---------------------------------------------------------------------------

typedef __bf16 bf16_t;
typedef __attribute__((ext_vector_type(16))) __bf16 bf16x16;
typedef __attribute__((ext_vector_type(8)))  __bf16 bf16x8;
typedef __attribute__((ext_vector_type(8)))  float  f32x8;

#define WMMA_BF16(A, B, C) \
  __builtin_amdgcn_wmma_f32_16x16x32_bf16(false, (A), false, (B), (short)0, (C), false, false)

// ---- TDM availability probing --------------------------------------------
// USE_TDM: 0 = vector-load fallback, 1 = 5-arg builtin (ROCm 7.2 clang-22),
//          2 = 6-arg builtin (amdgpu-toolchain clang-23 w/ TDM header).
#if defined(__has_builtin)
#if __has_builtin(__builtin_amdgcn_tensor_load_to_lds)
#if __has_include(<hip/amd_detail/amd_gfx1250_TDM.h>)
#define USE_TDM 2
#else
#define USE_TDM 1
#endif
#endif
#endif
#ifndef USE_TDM
#define USE_TDM 0
#endif

#if defined(__AMDGCN__)
#if USE_TDM == 1
#warning "CDNA5 device pass: TDM enabled via 5-arg builtin"
#elif USE_TDM == 2
#warning "CDNA5 device pass: TDM enabled via 6-arg builtin (TDM header present)"
#else
#warning "CDNA5 device pass: TDM builtin NOT found; using vector-load fallback"
#endif
#endif

#if defined(__has_builtin)
#if __has_builtin(__builtin_amdgcn_s_wait_tensorcnt)
#define HAVE_WAIT_TENSORCNT 1
#endif
#endif
#ifndef HAVE_WAIT_TENSORCNT
#define HAVE_WAIT_TENSORCNT 0
#endif

#if USE_TDM
typedef unsigned int u32x4 __attribute__((ext_vector_type(4)));
typedef int i32x4 __attribute__((ext_vector_type(4)));
typedef int i32x8 __attribute__((ext_vector_type(8)));

static __device__ inline void tdm_wait_le2() {
#if HAVE_WAIT_TENSORCNT
  __builtin_amdgcn_s_wait_tensorcnt(2);
#else
  asm volatile("s_wait_tensorcnt 0x2" ::: "memory");
#endif
}
static __device__ inline void tdm_wait_0() {
#if HAVE_WAIT_TENSORCNT
  __builtin_amdgcn_s_wait_tensorcnt(0);
#else
  asm volatile("s_wait_tensorcnt 0x0" ::: "memory");
#endif
}

// D# group 1 for a 2-D bf16 tile (ISA 8.4): data_size=2B, LDS padding enabled.
// pad after 2^pad_interval_log2 DWORDs stored, pad_dw DWORDs of padding.
static __device__ inline i32x8 tdm_desc_g1(int tensor_d0, int tensor_d1,
                                           long long stride0, int tile_d0,
                                           int tile_d1, int pad_interval_log2,
                                           int pad_dw) {
  i32x8 g;
  g[0] = (1 << 16) | (1 << 20) | ((pad_interval_log2 - 1) << 22) |
         ((pad_dw - 1) << 25);
  g[1] = (tensor_d0 & 0xffff) << 16;
  g[2] = ((tensor_d0 >> 16) & 0xffff) | ((tensor_d1 & 0xffff) << 16);
  g[3] = ((tensor_d1 >> 16) & 0xffff) | ((tile_d0 & 0xffff) << 16);
  g[4] = (tile_d1 & 0xffff);
  g[5] = (int)(stride0 & 0xffffffffll);
  g[6] = (int)((stride0 >> 32) & 0xffff);
  g[7] = 0;
  return g;
}

// D# group 0 (ISA 8.3): count=1, lds_addr, 57-bit global byte addr, type=2.
static __device__ inline void tdm_load_2d(unsigned lds_addr, const void* gaddr,
                                          i32x8 g1) {
  const unsigned long long ga = (unsigned long long)(uintptr_t)gaddr;
  u32x4 g0;
  g0[0] = 1u;
  g0[1] = lds_addr;
  g0[2] = (unsigned)ga;
  g0[3] = (unsigned)((ga >> 32) & 0x01ffffffu) | (2u << 30);
  i32x4 z = {0, 0, 0, 0};
#if USE_TDM == 2
  i32x8 z8 = {0, 0, 0, 0, 0, 0, 0, 0};
  __builtin_amdgcn_tensor_load_to_lds(g0, g1, z, z, z8, 0);
#else
  __builtin_amdgcn_tensor_load_to_lds(g0, g1, z, z, 0);
#endif
}
#endif  // USE_TDM

union FragAB { bf16x16 v; bf16x8 h[2]; };

// A-matrix 16x32 bf16 fragment (ISA 7.12.2): lane m<16 -> row m, K {0..7,16..23};
// lane m>=16 -> row m-16, K {8..15, 24..31}. base points at [row0][k0].
static __device__ inline bf16x16 load_frag_a(const bf16_t* base, int ld, int lane) {
  const bf16_t* p = base + (size_t)(lane & 15) * ld + ((lane & 16) ? 8 : 0);
  FragAB f;
  f.h[0] = *(const bf16x8*)(p);
  f.h[1] = *(const bf16x8*)(p + 16);
  return f.v;
}

// B-matrix 32x16 bf16 fragment from transposed tile BT[n][k] (k contiguous).
static __device__ inline bf16x16 load_frag_b(const bf16_t* base, int ld, int lane) {
  const bf16_t* p = base + (size_t)(lane & 15) * ld + ((lane & 16) ? 8 : 0);
  FragAB f;
  f.h[0] = *(const bf16x8*)(p);
  f.h[1] = *(const bf16x8*)(p + 16);
  return f.v;
}

// ---------------------------------------------------------------------------
// Conversion kernels
// ---------------------------------------------------------------------------
__global__ void cvt_f32_to_bf16(const float* __restrict__ in, bf16_t* __restrict__ out, int n) {
  int i = blockIdx.x * blockDim.x + threadIdx.x;
  if (i < n) out[i] = (bf16_t)in[i];
}

// W[K][N] fp32 -> WT[N][K] bf16
__global__ void transpose_w_to_bf16(const float* __restrict__ in, bf16_t* __restrict__ out,
                                    int K, int N) {
  int i = blockIdx.x * blockDim.x + threadIdx.x;
  if (i < K * N) {
    int k = i / N;
    int n = i - k * N;
    out[(size_t)n * K + k] = (bf16_t)in[i];
  }
}

// ---------------------------------------------------------------------------
// Tiled bf16 WMMA GEMM: C[M][N] = A[M][K] * BT[N][K]^T
// MODE 0: QKV epilogue (scatter Q*0.125 / K into [B,H,T,D], V into [B,H,D,T])
// MODE 1: fp32 epilogue: out[m][n] = acc + bias[n]
// Block tile 128x128x32, 256 threads = 8 waves; wave tile 32x64.
// TDM: double-buffered LDS, wave 0 prefetches tile k+1 while all compute k.
// ---------------------------------------------------------------------------
#define GBM 128
#define GBN 128
#define GBK 32
#define GLD 40  // LDS K-pitch: 32 elems (16 DW) + 4 DW pad = 40 elems

template <int MODE>
__global__ __launch_bounds__(256) void gemm_bf16_wmma(
    const bf16_t* __restrict__ A, const bf16_t* __restrict__ BT,
    const float* __restrict__ bias,
    bf16_t* __restrict__ outQ, bf16_t* __restrict__ outK, bf16_t* __restrict__ outV,
    float* __restrict__ outF, int M, int N, int K) {
  __shared__ bf16_t sA[2][GBM * GLD];
  __shared__ bf16_t sB[2][GBN * GLD];

  const int tid  = threadIdx.x;
  const int lane = tid & 31;
  const int wid  = tid >> 5;          // 0..7
  const int wm   = (wid & 3) * 32;    // wave M offset within tile
  const int wn   = (wid >> 2) * 64;   // wave N offset within tile
  const int m0   = blockIdx.y * GBM;
  const int n0   = blockIdx.x * GBN;

  f32x8 acc[2][4];
#pragma unroll
  for (int i = 0; i < 2; ++i)
#pragma unroll
    for (int j = 0; j < 4; ++j) acc[i][j] = (f32x8){0, 0, 0, 0, 0, 0, 0, 0};

  const int nkt = K / GBK;

#if USE_TDM
  // tensor = [rows][K] bf16, tile = GBK x 128 rows, pad 4 DW per 16 DW row.
  const i32x8 g1a = tdm_desc_g1(K, M, K, GBK, GBM, 4, 4);
  const i32x8 g1b = tdm_desc_g1(K, N, K, GBK, GBN, 4, 4);
  if (wid == 0) {
    tdm_load_2d((unsigned)(uintptr_t)&sA[0][0], A + (size_t)m0 * K, g1a);
    tdm_load_2d((unsigned)(uintptr_t)&sB[0][0], BT + (size_t)n0 * K, g1b);
  }
#else
  const int lrow = tid >> 1;          // 0..127
  const int lseg = (tid & 1) * 16;    // 0 or 16
#endif

  for (int kt = 0; kt < nkt; ++kt) {
    const int k0  = kt * GBK;
    const int cur = kt & 1;
#if USE_TDM
    if (wid == 0) {
      if (kt + 1 < nkt) {
        tdm_load_2d((unsigned)(uintptr_t)&sA[cur ^ 1][0],
                    A + (size_t)m0 * K + (k0 + GBK), g1a);
        tdm_load_2d((unsigned)(uintptr_t)&sB[cur ^ 1][0],
                    BT + (size_t)n0 * K + (k0 + GBK), g1b);
        tdm_wait_le2();  // previous tile's two loads complete (in-order)
      } else {
        tdm_wait_0();
      }
    }
#else
    {
      const bf16_t* ga = A + (size_t)(m0 + lrow) * K + k0 + lseg;
      *(bf16x8*)(&sA[cur][lrow * GLD + lseg])     = *(const bf16x8*)(ga);
      *(bf16x8*)(&sA[cur][lrow * GLD + lseg + 8]) = *(const bf16x8*)(ga + 8);
      const bf16_t* gb = BT + (size_t)(n0 + lrow) * K + k0 + lseg;
      *(bf16x8*)(&sB[cur][lrow * GLD + lseg])     = *(const bf16x8*)(gb);
      *(bf16x8*)(&sB[cur][lrow * GLD + lseg + 8]) = *(const bf16x8*)(gb + 8);
    }
#endif
    __syncthreads();

    bf16x16 af[2], bfv[4];
#pragma unroll
    for (int i = 0; i < 2; ++i)
      af[i] = load_frag_a(&sA[cur][(wm + i * 16) * GLD], GLD, lane);
#pragma unroll
    for (int j = 0; j < 4; ++j)
      bfv[j] = load_frag_b(&sB[cur][(wn + j * 16) * GLD], GLD, lane);
#pragma unroll
    for (int i = 0; i < 2; ++i)
#pragma unroll
      for (int j = 0; j < 4; ++j) acc[i][j] = WMMA_BF16(af[i], bfv[j], acc[i][j]);
    __syncthreads();
  }

  // Epilogue. C/D layout: lane, VGPR r -> row = r + (lane&16 ? 8:0), col = lane&15.
  const int ho = (lane & 16) ? 8 : 0;
  const int nl = lane & 15;
#pragma unroll
  for (int i = 0; i < 2; ++i) {
#pragma unroll
    for (int j = 0; j < 4; ++j) {
#pragma unroll
      for (int r = 0; r < 8; ++r) {
        const int gm = m0 + wm + i * 16 + r + ho;
        const int gn = n0 + wn + j * 16 + nl;
        float v = acc[i][j][r];
        if (MODE == 0) {
          const int b = gm >> 12;        // T = 4096
          const int t = gm & 4095;
          const int sel = gn / 768;      // 0=Q,1=K,2=V
          const int c = gn - sel * 768;
          const int h = c >> 6;
          const int d = c & 63;
          if (sel == 0) {
            outQ[(((size_t)(b * 12 + h) * 4096) + t) * 64 + d] = (bf16_t)(v * 0.125f);
          } else if (sel == 1) {
            outK[(((size_t)(b * 12 + h) * 4096) + t) * 64 + d] = (bf16_t)v;
          } else {
            outV[(((size_t)(b * 12 + h) * 64) + d) * 4096 + t] = (bf16_t)v;
          }
        } else {
          outF[(size_t)gm * N + gn] = v + bias[gn];
        }
      }
    }
  }
}

// ---------------------------------------------------------------------------
// Flash attention forward. Grid (T/64, B*H), 128 threads = 4 waves.
// Each wave owns a 16-row query tile; block streams 64-key K/V tiles via LDS
// (TDM double-buffered). Q pre-scaled by D^-0.5; K [B,H,T,D]; V [B,H,D,T].
// Output written bf16 [B,T,C] for the projection GEMM.
// ---------------------------------------------------------------------------
#define ALD 72  // LDS pitch: 64 elems (32 DW) + 4 DW pad = 72 elems

__global__ __launch_bounds__(128) void attn_fwd_wmma(
    const bf16_t* __restrict__ Q, const bf16_t* __restrict__ Kb,
    const bf16_t* __restrict__ Vb, bf16_t* __restrict__ Ob) {
  __shared__ bf16_t sK[2][64 * ALD];   // [key][d]
  __shared__ bf16_t sV[2][64 * ALD];   // [d][key]  (B^T for P*V)
  __shared__ bf16_t sP[4][16 * ALD];   // per-wave [qrow][key]

  const int tid   = threadIdx.x;
  const int lane  = tid & 31;
  const int wid   = tid >> 5;          // 0..3
  const int bh    = blockIdx.y;        // b*H + h
  const int b     = bh / 12;
  const int h     = bh - b * 12;
  const int qbase = blockIdx.x * 64;
  const int qrow0 = qbase + wid * 16;

  const bf16_t* Qh = Q  + (size_t)bh * 4096 * 64;
  const bf16_t* Kh = Kb + (size_t)bh * 4096 * 64;
  const bf16_t* Vh = Vb + (size_t)bh * 4096 * 64;

  // Loop-invariant Q fragments (two K=32 chunks over D=64), from global.
  bf16x16 qf[2];
  qf[0] = load_frag_a(Qh + (size_t)qrow0 * 64,      64, lane);
  qf[1] = load_frag_a(Qh + (size_t)qrow0 * 64 + 32, 64, lane);

  const int ho = (lane & 16) ? 8 : 0;
  const int nl = lane & 15;

  float m_st[8], l_st[8];
#pragma unroll
  for (int r = 0; r < 8; ++r) { m_st[r] = -1e30f; l_st[r] = 0.0f; }
  f32x8 o[4];
#pragma unroll
  for (int j = 0; j < 4; ++j) o[j] = (f32x8){0, 0, 0, 0, 0, 0, 0, 0};

  const int ntiles = (qbase + 64) >> 6;  // uniform across block

#if USE_TDM
  // K tensor [4096][64], tile 64x64; V tensor [64][4096], tile 64x64 at col kb.
  const i32x8 g1k = tdm_desc_g1(64, 4096, 64, 64, 64, 5, 4);
  const i32x8 g1v = tdm_desc_g1(4096, 64, 4096, 64, 64, 5, 4);
  if (wid == 0) {
    tdm_load_2d((unsigned)(uintptr_t)&sK[0][0], Kh, g1k);
    tdm_load_2d((unsigned)(uintptr_t)&sV[0][0], Vh, g1v);
  }
#else
  const int lrow = tid >> 1;           // 0..63
  const int lseg = (tid & 1) * 32;     // 0 or 32
#endif

  for (int it = 0; it < ntiles; ++it) {
    const int kb  = it << 6;
    const int cur = it & 1;
#if USE_TDM
    if (wid == 0) {
      if (it + 1 < ntiles) {
        tdm_load_2d((unsigned)(uintptr_t)&sK[cur ^ 1][0],
                    Kh + (size_t)(kb + 64) * 64, g1k);
        tdm_load_2d((unsigned)(uintptr_t)&sV[cur ^ 1][0], Vh + (kb + 64), g1v);
        tdm_wait_le2();
      } else {
        tdm_wait_0();
      }
    }
#else
    {
      const bf16_t* gk = Kh + (size_t)(kb + lrow) * 64 + lseg;
      const bf16_t* gv = Vh + (size_t)lrow * 4096 + kb + lseg;
#pragma unroll
      for (int s = 0; s < 4; ++s) {
        *(bf16x8*)(&sK[cur][lrow * ALD + lseg + s * 8]) = *(const bf16x8*)(gk + s * 8);
        *(bf16x8*)(&sV[cur][lrow * ALD + lseg + s * 8]) = *(const bf16x8*)(gv + s * 8);
      }
    }
#endif
    __syncthreads();

    // S = Q * K^T  (4 key subtiles x 2 K-chunks = 8 WMMAs)
    f32x8 s[4];
#pragma unroll
    for (int j = 0; j < 4; ++j) {
      f32x8 a = (f32x8){0, 0, 0, 0, 0, 0, 0, 0};
      a = WMMA_BF16(qf[0], load_frag_b(&sK[cur][(j * 16) * ALD],      ALD, lane), a);
      a = WMMA_BF16(qf[1], load_frag_b(&sK[cur][(j * 16) * ALD + 32], ALD, lane), a);
      s[j] = a;
    }

    // Causal mask + row max (reduce across the 16 lanes holding each row).
    float rmax[8];
#pragma unroll
    for (int r = 0; r < 8; ++r) rmax[r] = -1e30f;
#pragma unroll
    for (int j = 0; j < 4; ++j) {
      const int kk = kb + j * 16 + nl;
#pragma unroll
      for (int r = 0; r < 8; ++r) {
        const int qg = qrow0 + r + ho;
        float v = s[j][r];
        v = (kk > qg) ? -1e30f : v;
        s[j][r] = v;
        rmax[r] = fmaxf(rmax[r], v);
      }
    }
#pragma unroll
    for (int r = 0; r < 8; ++r) {
      float v = rmax[r];
      v = fmaxf(v, __shfl_xor(v, 1, 32));
      v = fmaxf(v, __shfl_xor(v, 2, 32));
      v = fmaxf(v, __shfl_xor(v, 4, 32));
      v = fmaxf(v, __shfl_xor(v, 8, 32));
      rmax[r] = v;
    }

    float alpha[8], rsum[8];
#pragma unroll
    for (int r = 0; r < 8; ++r) {
      const float mn = fmaxf(m_st[r], rmax[r]);
      alpha[r] = exp2f((m_st[r] - mn) * 1.44269504f);
      m_st[r] = mn;
      rsum[r] = 0.0f;
    }

    // P = exp2((S - m)*log2e); stage bf16 into per-wave LDS [row][key].
#pragma unroll
    for (int j = 0; j < 4; ++j) {
#pragma unroll
      for (int r = 0; r < 8; ++r) {
        const float p = exp2f((s[j][r] - m_st[r]) * 1.44269504f);
        rsum[r] += p;
        sP[wid][(r + ho) * ALD + j * 16 + nl] = (bf16_t)p;
      }
    }
#pragma unroll
    for (int r = 0; r < 8; ++r) {
      float v = rsum[r];
      v += __shfl_xor(v, 1, 32);
      v += __shfl_xor(v, 2, 32);
      v += __shfl_xor(v, 4, 32);
      v += __shfl_xor(v, 8, 32);
      l_st[r] = l_st[r] * alpha[r] + v;
    }
#pragma unroll
    for (int j = 0; j < 4; ++j)
#pragma unroll
      for (int r = 0; r < 8; ++r) o[j][r] *= alpha[r];

    __syncthreads();  // sP visible to the WMMA fragment reads

    // O += P * V  (4 d-subtiles x 2 key-chunks = 8 WMMAs)
#pragma unroll
    for (int j = 0; j < 4; ++j) {
      f32x8 a = o[j];
      a = WMMA_BF16(load_frag_a(&sP[wid][0],  ALD, lane),
                    load_frag_b(&sV[cur][(j * 16) * ALD],      ALD, lane), a);
      a = WMMA_BF16(load_frag_a(&sP[wid][32], ALD, lane),
                    load_frag_b(&sV[cur][(j * 16) * ALD + 32], ALD, lane), a);
      o[j] = a;
    }
    __syncthreads();  // all reads of sK/sV[cur] done before it is re-filled
  }

  // Normalize and write [B,T,C] bf16 (col = h*64 + d).
#pragma unroll
  for (int r = 0; r < 8; ++r) {
    const float inv = 1.0f / l_st[r];
    const int t = qrow0 + r + ho;
    const size_t rowp = ((size_t)b * 4096 + t) * 768 + (size_t)h * 64;
#pragma unroll
    for (int j = 0; j < 4; ++j) Ob[rowp + j * 16 + nl] = (bf16_t)(o[j][r] * inv);
  }
}

// ---------------------------------------------------------------------------
// Host-side orchestration
// ---------------------------------------------------------------------------
extern "C" void kernel_launch(void* const* d_in, const int* in_sizes, int n_in,
                              void* d_out, int out_size, void* d_ws, size_t ws_size,
                              hipStream_t stream) {
  const float* x     = (const float*)d_in[0];  // [2,4096,768]
  const float* Wqkv  = (const float*)d_in[1];  // [768,2304]
  const float* Wproj = (const float*)d_in[2];  // [768,768]
  const float* bproj = (const float*)d_in[3];  // [768]
  float* out = (float*)d_out;                  // [2,4096,768]

  const int Bn = 2, T = 4096, C = 768;
  const int M = Bn * T;  // 8192

  char* ws = (char*)d_ws;
  size_t off = 0;
  auto wsalloc = [&](size_t bytes) -> void* {
    void* p = ws + off;
    off += (bytes + 255) & ~(size_t)255;
    return p;
  };
  bf16_t* xb     = (bf16_t*)wsalloc((size_t)M * C * 2);
  bf16_t* wqkvT  = (bf16_t*)wsalloc((size_t)3 * C * C * 2);
  bf16_t* wprojT = (bf16_t*)wsalloc((size_t)C * C * 2);
  bf16_t* q      = (bf16_t*)wsalloc((size_t)M * C * 2);
  bf16_t* k      = (bf16_t*)wsalloc((size_t)M * C * 2);
  bf16_t* v      = (bf16_t*)wsalloc((size_t)M * C * 2);
  bf16_t* ao     = (bf16_t*)wsalloc((size_t)M * C * 2);

  {
    const int n = M * C;
    cvt_f32_to_bf16<<<(n + 255) / 256, 256, 0, stream>>>(x, xb, n);
  }
  {
    const int n = C * 3 * C;
    transpose_w_to_bf16<<<(n + 255) / 256, 256, 0, stream>>>(Wqkv, wqkvT, C, 3 * C);
  }
  {
    const int n = C * C;
    transpose_w_to_bf16<<<(n + 255) / 256, 256, 0, stream>>>(Wproj, wprojT, C, C);
  }

  // QKV projection: [8192 x 2304] = xb * WqkvT^T, scattered into Q/K/V.
  {
    dim3 grid(3 * C / GBN, M / GBM);
    gemm_bf16_wmma<0><<<grid, 256, 0, stream>>>(xb, wqkvT, nullptr, q, k, v, nullptr,
                                                M, 3 * C, C);
  }

  // Flash attention.
  {
    dim3 grid(T / 64, Bn * 12);
    attn_fwd_wmma<<<grid, 128, 0, stream>>>(q, k, v, ao);
  }

  // Output projection with bias -> fp32.
  {
    dim3 grid(C / GBN, M / GBM);
    gemm_bf16_wmma<1><<<grid, 256, 0, stream>>>(ao, wprojT, bproj, nullptr, nullptr,
                                                nullptr, out, M, C, C);
  }
}
